// Model_45114336477821
// MI455X (gfx1250) — compile-verified, matrix operand order checked by reading, not run on hardware
//
#include <hip/hip_runtime.h>
#include <hip/hip_bf16.h>
#include <math.h>

// ---------------------------------------------------------------------------
// Model dims
#define SEQL   30
#define DMOD   14
#define NHEAD  8
#define DKA    8      // H*E
#define DFF    50
#define MAK    25
#define NB     65536
#define KPAD   448    // 420 padded to 14*32 for WMMA K
// ---------------------------------------------------------------------------

typedef __attribute__((ext_vector_type(16))) _Float16 v16h;
typedef __attribute__((ext_vector_type(8)))  _Float16 v8h;
typedef __attribute__((ext_vector_type(8)))  float    v8f;

// Build a 16-half A/B WMMA fragment from two contiguous 8-half chunks.
// CDNA5 16-bit fragment layout: lanes 0-15 cover K {0..7,16..23} (+k0),
// lanes 16-31 cover K {8..15,24..31}; caller passes p already offset by
// row*stride + k0 + base, where base = (lane>=16 ? 8 : 0).
__device__ __forceinline__ v16h load_frag16(const _Float16* p) {
  v8h lo = *(const v8h*)p;
  v8h hi = *(const v8h*)(p + 16);
  v16h r;
#pragma unroll
  for (int i = 0; i < 8; ++i) { r[i] = lo[i]; r[i + 8] = hi[i]; }
  return r;
}

// ---------------------------------------------------------------------------
// Kernel W: adjacency softmax + f16 transposed weight prep
// W1T  : (256 x 448)  f16, K zero-padded 420->448
// WresT: (64  x 448)  f16
// W2T  : (64  x 256)  f16
// ---------------------------------------------------------------------------
__global__ void prep_kernel(const float* __restrict__ channels,
                            const float* __restrict__ W1,
                            const float* __restrict__ Wres,
                            const float* __restrict__ W2,
                            float* __restrict__ adj,
                            _Float16* __restrict__ W1T,
                            _Float16* __restrict__ WresT,
                            _Float16* __restrict__ W2T) {
  int idx = blockIdx.x * blockDim.x + threadIdx.x;
  int stride = gridDim.x * blockDim.x;

  for (int c = idx; c < DMOD; c += stride) {
    float mx = -1e30f;
    for (int j = 0; j < DMOD; ++j) mx = fmaxf(mx, channels[c * DMOD + j]);
    float e[DMOD], s = 0.f;
    for (int j = 0; j < DMOD; ++j) { e[j] = expf(channels[c * DMOD + j] - mx); s += e[j]; }
    float inv = 1.f / s;
    for (int j = 0; j < DMOD; ++j) adj[c * DMOD + j] = e[j] * inv;
  }
  for (int i = idx; i < 256 * KPAD; i += stride) {
    int n = i / KPAD, k = i % KPAD;
    W1T[i] = (k < 420) ? (_Float16)W1[k * 256 + n] : (_Float16)0.f;
  }
  for (int i = idx; i < 64 * KPAD; i += stride) {
    int n = i / KPAD, k = i % KPAD;
    WresT[i] = (k < 420) ? (_Float16)Wres[k * 64 + n] : (_Float16)0.f;
  }
  for (int i = idx; i < 64 * 256; i += stride) {
    int n = i / 256, k = i % 256;
    W2T[i] = (_Float16)W2[k * 64 + n];
  }
}

// ---------------------------------------------------------------------------
// Kernel A: per-sample front end. One wave32 per sample, 2 waves per block.
// Produces F16 (B x 448) feature rows for the WMMA head.
// Arena per wave (floats): XA 420 | XB 420 | Q 240 | K 240 | V 240 | AG 240 |
//                          HID 1504 | MC 32 | SC 24   => 3360
// ---------------------------------------------------------------------------
#define ARENA 3360

__global__ __launch_bounds__(64) void frontend_kernel(
    const float* __restrict__ x,
    const float* __restrict__ W_emb, const float* __restrict__ b_emb,
    const float* __restrict__ Wq, const float* __restrict__ bq,
    const float* __restrict__ Wk, const float* __restrict__ bk,
    const float* __restrict__ Wv, const float* __restrict__ bv,
    const float* __restrict__ Wo, const float* __restrict__ bo,
    const float* __restrict__ W_ff1, const float* __restrict__ W_ff2,
    const float* __restrict__ g_norm, const float* __restrict__ b_norm,
    const float* __restrict__ W_dy, const float* __restrict__ b_dy,
    const float* __restrict__ adj,
    _Float16* __restrict__ F16) {
  __shared__ float arena[2 * ARENA];
  const int wave = threadIdx.x >> 5;
  const int lane = threadIdx.x & 31;
  const int s = blockIdx.x * 2 + wave;

  float* A   = arena + wave * ARENA;
  float* XA  = A;            // 420
  float* XB  = A + 420;      // 420
  float* Qb  = A + 840;      // 240 (8x30)
  float* Kb  = A + 1080;     // 240
  float* Vb  = A + 1320;     // 240
  float* AG  = A + 1560;     // 240
  float* HID = A + 1800;     // 1504
  float* MC  = A + 3304;     // 32
  float* SC  = A + 3336;     // 24

  // 1) load x (L,D)
  for (int i = lane; i < 420; i += 32) XA[i] = x[(size_t)s * 420 + i];
  __syncthreads();

  // 2) conv1d embedding (zero pad 1), out -> XB (== xe)
  for (int i = lane; i < 420; i += 32) {
    int t = i / DMOD, c = i % DMOD;
    float acc = b_emb[c];
#pragma unroll
    for (int kk = 0; kk < 3; ++kk) {
      int tt = t + kk - 1;
      if (tt < 0 || tt >= SEQL) continue;
      for (int ci = 0; ci < DMOD; ++ci)
        acc += W_emb[c * (DMOD * 3) + ci * 3 + kk] * XA[tt * DMOD + ci];
    }
    XB[i] = acc;
  }
  __syncthreads();

  // 3) q/k/v: (L,14)@ (14,8) -> stored (h,t)
  for (int i = lane; i < NHEAD * SEQL; i += 32) {
    int h = i / SEQL, t = i % SEQL;
    float aq = bq[h], ak = bk[h], av = bv[h];
    for (int c = 0; c < DMOD; ++c) {
      float xv = XB[t * DMOD + c];
      aq += xv * Wq[c * DKA + h];
      ak += xv * Wk[c * DKA + h];
      av += xv * Wv[c * DKA + h];
    }
    Qb[i] = aq; Kb[i] = ak; Vb[i] = av;
  }
  __syncthreads();

  // 4) circular autocorrelation, mean over heads: corr[tau]=sum q[(t+tau)%L]k[t]
  if (lane < SEQL) {
    int tau = lane;
    float acc = 0.f;
    for (int h = 0; h < NHEAD; ++h)
      for (int t = 0; t < SEQL; ++t)
        acc += Qb[h * SEQL + ((t + tau) % SEQL)] * Kb[h * SEQL + t];
    MC[tau] = acc * (1.0f / NHEAD);
  }
  __syncthreads();

  // 5) top-3 + softmax (lane 0, serial)
  if (lane == 0) {
    int u0 = -1, u1 = -1;
    float w[3]; int d[3];
    for (int j = 0; j < 3; ++j) {
      float best = -1e30f; int bi = 0;
      for (int t = 0; t < SEQL; ++t) {
        if (t == u0 || t == u1) continue;
        if (MC[t] > best) { best = MC[t]; bi = t; }
      }
      w[j] = best; d[j] = bi;
      if (j == 0) u0 = bi; else if (j == 1) u1 = bi;
    }
    float e0 = expf(w[0] - w[0]), e1 = expf(w[1] - w[0]), e2 = expf(w[2] - w[0]);
    float inv = 1.f / (e0 + e1 + e2);
    SC[0] = e0 * inv; SC[1] = e1 * inv; SC[2] = e2 * inv;
    SC[3] = (float)d[0]; SC[4] = (float)d[1]; SC[5] = (float)d[2];
  }
  __syncthreads();

  // 6) delay aggregation
  {
    float w0 = SC[0], w1 = SC[1], w2 = SC[2];
    int d0 = (int)SC[3], d1 = (int)SC[4], d2 = (int)SC[5];
    for (int i = lane; i < NHEAD * SEQL; i += 32) {
      int h = i / SEQL, t = i % SEQL;
      AG[i] = w0 * Vb[h * SEQL + ((t + d0) % SEQL)] +
              w1 * Vb[h * SEQL + ((t + d1) % SEQL)] +
              w2 * Vb[h * SEQL + ((t + d2) % SEQL)];
    }
  }
  __syncthreads();

  // 7) x = xe + agg @ Wo + bo  -> XA
  for (int i = lane; i < 420; i += 32) {
    int t = i / DMOD, c = i % DMOD;
    float acc = XB[i] + bo[c];
    for (int h = 0; h < NHEAD; ++h) acc += AG[h * SEQL + t] * Wo[h * DMOD + c];
    XA[i] = acc;
  }
  __syncthreads();

  // 8) series_decomp #1: XB = XA - movavg25(replicate-pad)
  for (int i = lane; i < 420; i += 32) {
    int t = i / DMOD, c = i % DMOD;
    float acc = 0.f;
    for (int j = -12; j <= 12; ++j) {
      int tt = t + j; tt = tt < 0 ? 0 : (tt > SEQL - 1 ? SEQL - 1 : tt);
      acc += XA[tt * DMOD + c];
    }
    XB[i] = XA[i] - acc * (1.0f / MAK);
  }
  __syncthreads();

  // 9) FFN hidden = gelu(x @ W_ff1) (exact erf gelu)
  for (int i = lane; i < SEQL * DFF; i += 32) {
    int t = i / DFF, dd = i % DFF;
    float a = 0.f;
    for (int c = 0; c < DMOD; ++c) a += XB[t * DMOD + c] * W_ff1[c * DFF + dd];
    HID[i] = 0.5f * a * (1.0f + erff(a * 0.70710678118654752f));
  }
  __syncthreads();

  // 10) x2 = x + hidden @ W_ff2 -> XA
  for (int i = lane; i < 420; i += 32) {
    int t = i / DMOD, c = i % DMOD;
    float a = XB[i];
    for (int dd = 0; dd < DFF; ++dd) a += HID[t * DFF + dd] * W_ff2[dd * DMOD + c];
    XA[i] = a;
  }
  __syncthreads();

  // 11) series_decomp #2 -> XB
  for (int i = lane; i < 420; i += 32) {
    int t = i / DMOD, c = i % DMOD;
    float acc = 0.f;
    for (int j = -12; j <= 12; ++j) {
      int tt = t + j; tt = tt < 0 ? 0 : (tt > SEQL - 1 ? SEQL - 1 : tt);
      acc += XA[tt * DMOD + c];
    }
    XB[i] = XA[i] - acc * (1.0f / MAK);
  }
  __syncthreads();

  // 12) layernorm over channels
  if (lane < SEQL) {
    int t = lane;
    float m = 0.f;
    for (int c = 0; c < DMOD; ++c) m += XB[t * DMOD + c];
    m *= (1.0f / DMOD);
    float v = 0.f;
    for (int c = 0; c < DMOD; ++c) { float d = XB[t * DMOD + c] - m; v += d * d; }
    v *= (1.0f / DMOD);
    float inv = rsqrtf(v + 1e-5f);
    for (int c = 0; c < DMOD; ++c)
      XB[t * DMOD + c] = (XB[t * DMOD + c] - m) * inv * g_norm[c] + b_norm[c];
  }
  __syncthreads();

  // 13) DyConv time-linear: h[c][t'] = sum_t x[t][c] W_dy[t][t'] + b_dy[t']
  for (int i = lane; i < DMOD * SEQL; i += 32) {
    int c = i / SEQL, tp = i % SEQL;
    float a = b_dy[tp];
    for (int t = 0; t < SEQL; ++t) a += XB[t * DMOD + c] * W_dy[t * SEQL + tp];
    HID[i] = a;
  }
  __syncthreads();

  // 14) dy = relu(adj @ h), write f16 feature row (channel-major, pad to 448)
  for (int i = lane; i < KPAD; i += 32) {
    _Float16 hv = (_Float16)0.f;
    if (i < 420) {
      int c = i / SEQL, t = i % SEQL;
      float a = 0.f;
      for (int j = 0; j < DMOD; ++j) a += adj[c * DMOD + j] * HID[j * SEQL + t];
      hv = (_Float16)(a > 0.f ? a : 0.f);
    }
    F16[(size_t)s * KPAD + i] = hv;
  }
}

// ---------------------------------------------------------------------------
// Kernel B: WMMA head.  out = LN(relu(F@W1+b1)@W2 + b2 + F@Wres + bres) @ W3 + b3
// 4 waves / block, each wave owns a 16-row block of F. No inter-wave sync.
// Software-pipelined: next A fragment and next B-fragment chunk are issued
// before the current chunk's WMMAs, so load waits become partial.
// ---------------------------------------------------------------------------
__global__ __launch_bounds__(128) void head_kernel(
    const _Float16* __restrict__ F16,
    const _Float16* __restrict__ W1T, const _Float16* __restrict__ WresT,
    const _Float16* __restrict__ W2T,
    const float* __restrict__ b1, const float* __restrict__ b2,
    const float* __restrict__ bres,
    const float* __restrict__ g_ln, const float* __restrict__ b_ln,
    const float* __restrict__ W3, const float* __restrict__ b3,
    float* __restrict__ out) {
  __shared__ __align__(16) _Float16 lds[4 * 16 * 256];  // 8 KB / wave
  const int wave = threadIdx.x >> 5;
  const int lane = threadIdx.x & 31;
  const int m0 = (blockIdx.x * 4 + wave) * 16;
  _Float16* c1 = lds + wave * (16 * 256);

  const int rA = lane & 15;            // fragment row / column within tile
  const int base = (lane >> 4) * 8;    // K-half offset for this lane group
  const int rowBase = (lane >> 4) * 8; // C/D fragment row base

  const _Float16* pa_base = F16 + (size_t)(m0 + rA) * KPAD + base;

  __builtin_prefetch(W1T, 0, 3);
  __builtin_prefetch(WresT, 0, 3);
  __builtin_prefetch(W2T, 0, 3);

  // ---- GEMM1: C1(16x256) = F(16x448) @ W1 ----
  v8f acc[16];
#pragma unroll
  for (int nt = 0; nt < 16; ++nt) acc[nt] = (v8f)0.0f;

  {
    v16h a = load_frag16(pa_base);
#pragma unroll 1
    for (int k0 = 0; k0 < KPAD / 32; ++k0) {
      // chunk 0 B fragments
      v16h bbuf[4];
#pragma unroll
      for (int j = 0; j < 4; ++j)
        bbuf[j] = load_frag16(W1T + (j * 16 + rA) * KPAD + k0 * 32 + base);
      // prefetch next-k A fragment early
      v16h a_next = a;
      if (k0 < KPAD / 32 - 1) a_next = load_frag16(pa_base + (k0 + 1) * 32);
#pragma unroll
      for (int c = 0; c < 4; ++c) {
        v16h bn[4];
        if (c < 3) {
#pragma unroll
          for (int j = 0; j < 4; ++j)
            bn[j] = load_frag16(W1T + (((c + 1) * 4 + j) * 16 + rA) * KPAD + k0 * 32 + base);
        }
#pragma unroll
        for (int j = 0; j < 4; ++j)
          acc[c * 4 + j] = __builtin_amdgcn_wmma_f32_16x16x32_f16(
              false, a, false, bbuf[j], (short)0, acc[c * 4 + j], false, false);
        if (c < 3) {
#pragma unroll
          for (int j = 0; j < 4; ++j) bbuf[j] = bn[j];
        }
      }
      a = a_next;
    }
  }

  // relu + b1, stage C1 to LDS as f16 (16 x 256, row-major)
#pragma unroll
  for (int nt = 0; nt < 16; ++nt) {
    int col = nt * 16 + rA;
    float bias = b1[col];
#pragma unroll
    for (int r = 0; r < 8; ++r) {
      float v = acc[nt][r] + bias;
      c1[(rowBase + r) * 256 + col] = (_Float16)(v > 0.f ? v : 0.f);
    }
  }
  asm volatile("s_wait_dscnt 0" ::: "memory");

  // ---- GEMM2 + GEMM3 into shared 16x64 accumulators ----
  v8f acc2[4];
#pragma unroll
  for (int nt = 0; nt < 4; ++nt) acc2[nt] = (v8f)0.0f;

  // R = F @ Wres   (double-buffered across K)
  {
    v16h a = load_frag16(pa_base);
    v16h bb[4];
#pragma unroll
    for (int j = 0; j < 4; ++j)
      bb[j] = load_frag16(WresT + (j * 16 + rA) * KPAD + base);
#pragma unroll 1
    for (int k0 = 0; k0 < KPAD / 32; ++k0) {
      v16h an = a;
      v16h bn[4];
      if (k0 < KPAD / 32 - 1) {
        an = load_frag16(pa_base + (k0 + 1) * 32);
#pragma unroll
        for (int j = 0; j < 4; ++j)
          bn[j] = load_frag16(WresT + (j * 16 + rA) * KPAD + (k0 + 1) * 32 + base);
      }
#pragma unroll
      for (int j = 0; j < 4; ++j)
        acc2[j] = __builtin_amdgcn_wmma_f32_16x16x32_f16(
            false, a, false, bb[j], (short)0, acc2[j], false, false);
      if (k0 < KPAD / 32 - 1) {
        a = an;
#pragma unroll
        for (int j = 0; j < 4; ++j) bb[j] = bn[j];
      }
    }
  }
  // += C1 @ W2   (A fragments read back from LDS, double-buffered)
  {
    v16h a = load_frag16(c1 + rA * 256 + base);
    v16h bb[4];
#pragma unroll
    for (int j = 0; j < 4; ++j)
      bb[j] = load_frag16(W2T + (j * 16 + rA) * 256 + base);
#pragma unroll 1
    for (int k0 = 0; k0 < 256 / 32; ++k0) {
      v16h an = a;
      v16h bn[4];
      if (k0 < 256 / 32 - 1) {
        an = load_frag16(c1 + rA * 256 + (k0 + 1) * 32 + base);
#pragma unroll
        for (int j = 0; j < 4; ++j)
          bn[j] = load_frag16(W2T + (j * 16 + rA) * 256 + (k0 + 1) * 32 + base);
      }
#pragma unroll
      for (int j = 0; j < 4; ++j)
        acc2[j] = __builtin_amdgcn_wmma_f32_16x16x32_f16(
            false, a, false, bb[j], (short)0, acc2[j], false, false);
      if (k0 < 256 / 32 - 1) {
        a = an;
#pragma unroll
        for (int j = 0; j < 4; ++j) bb[j] = bn[j];
      }
    }
  }

  // bias (b2 + bres), stage 16x64 f32 tile to LDS (reuse c1 arena)
  float* lnb = (float*)c1;
#pragma unroll
  for (int nt = 0; nt < 4; ++nt) {
    int col = nt * 16 + rA;
    float bias = b2[col] + bres[col];
#pragma unroll
    for (int r = 0; r < 8; ++r)
      lnb[(rowBase + r) * 64 + col] = acc2[nt][r] + bias;
  }
  asm volatile("s_wait_dscnt 0" ::: "memory");

  // ---- per-row LayerNorm(64) then dot with W3 ----
  if (lane < 16) {
    const float* pr = lnb + lane * 64;
    float m = 0.f;
    for (int n = 0; n < 64; ++n) m += pr[n];
    m *= (1.0f / 64.0f);
    float var = 0.f;
    for (int n = 0; n < 64; ++n) { float d = pr[n] - m; var += d * d; }
    var *= (1.0f / 64.0f);
    float inv = rsqrtf(var + 1e-5f);
    float dot = 0.f;
    for (int n = 0; n < 64; ++n)
      dot += ((pr[n] - m) * inv * g_ln[n] + b_ln[n]) * W3[n];
    out[m0 + lane] = dot + b3[0];
  }
}

// ---------------------------------------------------------------------------
extern "C" void kernel_launch(void* const* d_in, const int* in_sizes, int n_in,
                              void* d_out, int out_size, void* d_ws, size_t ws_size,
                              hipStream_t stream) {
  (void)in_sizes; (void)n_in; (void)out_size; (void)ws_size;
  const float* x      = (const float*)d_in[0];
  const float* W_emb  = (const float*)d_in[1];
  const float* b_emb  = (const float*)d_in[2];
  const float* Wq     = (const float*)d_in[3];
  const float* bq     = (const float*)d_in[4];
  const float* Wk     = (const float*)d_in[5];
  const float* bk     = (const float*)d_in[6];
  const float* Wv     = (const float*)d_in[7];
  const float* bv     = (const float*)d_in[8];
  const float* Wo     = (const float*)d_in[9];
  const float* bo     = (const float*)d_in[10];
  const float* W_ff1  = (const float*)d_in[11];
  const float* W_ff2  = (const float*)d_in[12];
  const float* g_norm = (const float*)d_in[13];
  const float* b_norm = (const float*)d_in[14];
  const float* W_dy   = (const float*)d_in[15];
  const float* b_dy   = (const float*)d_in[16];
  const float* chans  = (const float*)d_in[17];
  const float* W1     = (const float*)d_in[18];
  const float* b1     = (const float*)d_in[19];
  const float* W2     = (const float*)d_in[20];
  const float* b2     = (const float*)d_in[21];
  const float* Wres   = (const float*)d_in[22];
  const float* bres   = (const float*)d_in[23];
  const float* g_ln   = (const float*)d_in[24];
  const float* b_ln   = (const float*)d_in[25];
  const float* W3     = (const float*)d_in[26];
  const float* b3     = (const float*)d_in[27];

  char* ws = (char*)d_ws;
  size_t off = 0;
  _Float16* F16 = (_Float16*)(ws + off);   off += (size_t)NB * KPAD * sizeof(_Float16);
  _Float16* W1T = (_Float16*)(ws + off);   off += (size_t)256 * KPAD * sizeof(_Float16);
  _Float16* WresT = (_Float16*)(ws + off); off += (size_t)64 * KPAD * sizeof(_Float16);
  _Float16* W2T = (_Float16*)(ws + off);   off += (size_t)64 * 256 * sizeof(_Float16);
  float* adj = (float*)(ws + off);         off += (size_t)DMOD * DMOD * sizeof(float);

  prep_kernel<<<512, 256, 0, stream>>>(chans, W1, Wres, W2, adj, W1T, WresT, W2T);

  frontend_kernel<<<NB / 2, 64, 0, stream>>>(
      x, W_emb, b_emb, Wq, bq, Wk, bk, Wv, bv, Wo, bo, W_ff1, W_ff2,
      g_norm, b_norm, W_dy, b_dy, adj, F16);

  head_kernel<<<NB / (4 * 16), 128, 0, stream>>>(
      F16, W1T, WresT, W2T, b1, b2, bres, g_ln, b_ln, W3, b3, (float*)d_out);
}
